// HeteroGNN_46462956208314
// MI455X (gfx1250) — matrix operand order repaired and checked by reading, not compiled, not run on hardware
//
#include <hip/hip_runtime.h>

#define N_NODES 50000
#define N_EDGES 400000
#define NREL    35
#define NBASIS  12
#define CDIM    128
#define NTYPE   4
#define HLD     38
#define KPAD    96   // real K = NREL*2 = 70, padded to 3 x 32 for wmma f16 K-chunks
#define NROWT   (N_NODES / 16)   // 3125 row tiles

typedef __attribute__((ext_vector_type(16))) _Float16 v16h;
typedef __attribute__((ext_vector_type(8)))  float    v8f;

// ---- float atomic max via ordered int-bits trick (init must be -inf) ----
__device__ __forceinline__ void atomicMaxF(float* addr, float val) {
  if (val >= 0.f) atomicMax((int*)addr, __float_as_int(val));
  else            atomicMin((unsigned int*)addr, (unsigned int)__float_as_int(val));
}

// ---- init workspace accumulators (ws is poisoned; re-init every call) ----
__global__ void k_init(float* __restrict__ amax, float* __restrict__ den,
                       float* __restrict__ s) {
  int i = blockIdx.x * blockDim.x + threadIdx.x;
  if (i < N_NODES) { amax[i] = -__builtin_huge_valf(); den[i] = 0.f; }
  if (i < N_NODES * KPAD) s[i] = 0.f;
}

// ---- w[r,i,c] = sum_b att_rel[r,b]*basis[b,i,c]
//      bfrag: f16 B-matrix pre-swizzled to WMMA fragment order:
//      logical B[k][c] -> bfrag[ ((k>>5)*2 + ((k>>3)&1))*2048 + c*16 + (k&7) + (((k>>4)&1)<<3) ]
//      so a lane's 16 fragment elements are 32 contiguous bytes. Rows 70..95 = 0. ----
__global__ void k_compose_w(const float* __restrict__ att_rel,
                            const float* __restrict__ basis,
                            float* __restrict__ wfull, _Float16* __restrict__ bfrag) {
  int idx = blockIdx.x * blockDim.x + threadIdx.x;   // over KPAD*CDIM
  if (idx >= KPAD * CDIM) return;
  int k = idx / CDIM, c = idx % CDIM;
  float acc = 0.f;
  if (k < NREL * 2) {
    int r = k >> 1, i = k & 1;
    #pragma unroll
    for (int b = 0; b < NBASIS; ++b)
      acc += att_rel[r * NBASIS + b] * basis[(b * 2 + i) * CDIM + c];
    wfull[idx] = acc;
  }
  int fi = ((k >> 5) * 2 + ((k >> 3) & 1)) * 2048 + c * 16 + (k & 7) + (((k >> 4) & 1) << 3);
  bfrag[fi] = (_Float16)acc;
}

// ---- wq[r,i] = w[r,i,:]@q_att ; wk likewise ; ew[i] = lin_edge_W[i,:]@e_att ----
__global__ void k_compose_small(const float* __restrict__ wfull,
                                const float* __restrict__ q, const float* __restrict__ kv,
                                const float* __restrict__ linW, const float* __restrict__ eatt,
                                float* __restrict__ wq, float* __restrict__ wk,
                                float* __restrict__ ew) {
  int tid = threadIdx.x;
  if (tid < NREL * 2) {
    float aq = 0.f, ak = 0.f;
    const float* wp = wfull + (size_t)tid * CDIM;
    for (int c = 0; c < CDIM; ++c) { aq += wp[c] * q[c]; ak += wp[c] * kv[c]; }
    wq[tid] = aq; wk[tid] = ak;
  } else if (tid < NREL * 2 + 2) {
    int i = tid - NREL * 2;
    float a = 0.f;
    for (int c = 0; c < CDIM; ++c) a += linW[i * CDIM + c] * eatt[c];
    ew[i] = a;
  }
}

// ---- per-edge attention logit (leaky relu) + segment max via atomics ----
__global__ void k_edge_alpha(const float* __restrict__ x, const int* __restrict__ ei,
                             const int* __restrict__ et, const float* __restrict__ ea,
                             const float* __restrict__ wq, const float* __restrict__ wk,
                             const float* __restrict__ ew,
                             float* __restrict__ alpha, float* __restrict__ amax) {
  int e = blockIdx.x * blockDim.x + threadIdx.x;
  if (e >= N_EDGES) return;
  int src = ei[e], dst = ei[N_EDGES + e], r = et[e];
  float a = x[dst * 2]     * wq[r * 2]     + x[dst * 2 + 1] * wq[r * 2 + 1]
          + x[src * 2]     * wk[r * 2]     + x[src * 2 + 1] * wk[r * 2 + 1]
          + ea[e * 2] * ew[0] + ea[e * 2 + 1] * ew[1];
  a = a > 0.f ? a : 0.2f * a;
  alpha[e] = a;
  atomicMaxF(&amax[dst], a);
}

// ---- exp(alpha - max) + segment-sum denominator ----
__global__ void k_edge_exp(const int* __restrict__ ei, float* __restrict__ alpha,
                           const float* __restrict__ amax, float* __restrict__ den) {
  int e = blockIdx.x * blockDim.x + threadIdx.x;
  if (e >= N_EDGES) return;
  int dst = ei[N_EDGES + e];
  float ex = __expf(alpha[e] - amax[dst]);
  alpha[e] = ex;
  atomicAdd(&den[dst], ex);
}

// ---- scatter normalized coeffs into s[n, r*2+i] (A-matrix of the big GEMM) ----
__global__ void k_edge_scatter(const float* __restrict__ x, const int* __restrict__ ei,
                               const int* __restrict__ et, const float* __restrict__ alpha,
                               const float* __restrict__ den, float* __restrict__ s) {
  int e = blockIdx.x * blockDim.x + threadIdx.x;
  if (e >= N_EDGES) return;
  int src = ei[e], dst = ei[N_EDGES + e], r = et[e];
  float an = alpha[e] / (den[dst] + 1e-16f);
  atomicAdd(&s[(size_t)dst * KPAD + r * 2],     an * x[src * 2]);
  atomicAdd(&s[(size_t)dst * KPAD + r * 2 + 1], an * x[src * 2 + 1]);
}

// ---- h = relu( S[N,KPAD] x B[KPAD,C] + bias ) via v_wmma_f32_16x16x32_f16.
//      B staged once per block in LDS (fragment order); each wave owns one
//      16-row tile, loads its 3 A-fragments once, loops 8 column tiles:
//      24 wmma per wave, S read exactly once from HBM. ----
__global__ void k_gemm_wmma(const float* __restrict__ S, const _Float16* __restrict__ Bfrag,
                            const float* __restrict__ bias, float* __restrict__ H) {
  __shared__ _Float16 Bs[KPAD * CDIM];           // 12288 halfs = 24 KB
  int tid = threadIdx.x;
  // cooperative 16-byte staging: 12288 halfs = 1536 float4 chunks / 256 threads
  #pragma unroll
  for (int i = 0; i < (KPAD * CDIM) / (8 * 256); ++i)
    ((float4*)Bs)[tid + i * 256] = ((const float4*)Bfrag)[tid + i * 256];
  __syncthreads();

  int rowTile = blockIdx.x * 8 + (tid >> 5);
  if (rowTile >= NROWT) return;                  // wave-uniform: EXEC all-ones for wmma
  int lane = tid & 31, g = lane >> 4, r = lane & 15;
  int row0 = rowTile * 16;

  // A fragments (16x32 f16): row M = lane&15; half g holds K {0..7,16..23}+8g
  v16h a[3];
  const float* Srow = S + (size_t)(row0 + r) * KPAD;
  #pragma unroll
  for (int ck = 0; ck < 3; ++ck) {
    const float* Sp = Srow + ck * 32 + g * 8;
    #pragma unroll
    for (int j = 0; j < 16; ++j) {
      int v = j >> 1, p = j & 1;
      int kl = (v < 4) ? (2 * v + p) : (16 + 2 * (v - 4) + p);
      a[ck][j] = (_Float16)Sp[kl];
    }
  }

  #pragma unroll
  for (int ct = 0; ct < 8; ++ct) {
    int col = ct * 16 + r;
    v8f c = {};
    #pragma unroll
    for (int ck = 0; ck < 3; ++ck) {
      // fragment-order LDS read: 16 contiguous halfs (32 B) -> 2x ds_load_b128
      v16h b = *(const v16h*)(Bs + (ck * 2 + g) * 2048 + col * 16);
      c = __builtin_amdgcn_wmma_f32_16x16x32_f16(false, a[ck], false, b,
                                                 (short)0, c, false, false);
    }
    float bc = bias[col];
    #pragma unroll
    for (int v = 0; v < 8; ++v) {                // C/D: M=v / v+8 per lane half, N=lane&15
      int m = (lane < 16) ? v : v + 8;
      float val = c[v] + bc;                     // relu(leaky_relu(x,.2)) == relu(x)
      H[(size_t)(row0 + m) * CDIM + col] = val > 0.f ? val : 0.f;
    }
  }
}

// ---- per-node-type MLP: 128->38->38->38->2, abs on out[1] for type 0 ----
__global__ void k_mlp(const float* __restrict__ H, const int* __restrict__ ntype,
                      const float* __restrict__ W0, const float* __restrict__ b0,
                      const float* __restrict__ W1, const float* __restrict__ b1,
                      const float* __restrict__ W2, const float* __restrict__ b2,
                      const float* __restrict__ Wf, const float* __restrict__ bf,
                      float* __restrict__ out) {
  __shared__ float hs[64 * 129];   // stride 129: avoid 64-bank conflicts
  __shared__ float us[64 * 39];    // stride 39 (coprime with 64)
  __shared__ float vs[64 * 39];
  int tid = threadIdx.x;
  int n = blockIdx.x * 64 + tid;
  if (n >= N_NODES) return;        // threads touch only their own LDS rows; no syncs needed
  int t = ntype[n];

  float* hrow = hs + tid * 129;
  const float* hp = H + (size_t)n * CDIM;
  for (int d = 0; d < CDIM; d += 4) {
    float4 v4 = *(const float4*)(hp + d);
    hrow[d] = v4.x; hrow[d + 1] = v4.y; hrow[d + 2] = v4.z; hrow[d + 3] = v4.w;
  }
  float* u = us + tid * 39;
  float* w = vs + tid * 39;

  const float* Wt = W0 + (size_t)t * CDIM * HLD;
  const float* bt = b0 + t * HLD;
  for (int o = 0; o < HLD; ++o) {
    float acc = bt[o];
    for (int d = 0; d < CDIM; ++d) acc += hrow[d] * Wt[d * HLD + o];
    u[o] = acc;
  }
  Wt = W1 + (size_t)t * HLD * HLD; bt = b1 + t * HLD;
  for (int o = 0; o < HLD; ++o) {
    float acc = bt[o];
    for (int d = 0; d < HLD; ++d) { float xv = u[d]; xv = xv > 0.f ? xv : 0.f; acc += xv * Wt[d * HLD + o]; }
    w[o] = acc;
  }
  Wt = W2 + (size_t)t * HLD * HLD; bt = b2 + t * HLD;
  for (int o = 0; o < HLD; ++o) {
    float acc = bt[o];
    for (int d = 0; d < HLD; ++d) { float xv = w[d]; xv = xv > 0.f ? xv : 0.f; acc += xv * Wt[d * HLD + o]; }
    u[o] = acc;
  }
  Wt = Wf + (size_t)t * HLD * 2; bt = bf + t * 2;
  float o0 = bt[0], o1 = bt[1];
  for (int d = 0; d < HLD; ++d) { o0 += u[d] * Wt[d * 2]; o1 += u[d] * Wt[d * 2 + 1]; }
  if (t == 0) o1 = fabsf(o1);
  out[n * 2] = o0; out[n * 2 + 1] = o1;
}

extern "C" void kernel_launch(void* const* d_in, const int* in_sizes, int n_in,
                              void* d_out, int out_size, void* d_ws, size_t ws_size,
                              hipStream_t stream) {
  const float* x     = (const float*)d_in[0];
  const int*   ei    = (const int*)  d_in[1];
  const int*   et    = (const int*)  d_in[2];
  const float* ea    = (const float*)d_in[3];
  const int*   nty   = (const int*)  d_in[4];
  const float* basis = (const float*)d_in[5];
  const float* att   = (const float*)d_in[6];
  const float* qv    = (const float*)d_in[7];
  const float* kv    = (const float*)d_in[8];
  const float* ev    = (const float*)d_in[9];
  const float* lew   = (const float*)d_in[10];
  const float* cbias = (const float*)d_in[11];
  const float* W0 = (const float*)d_in[12]; const float* b0 = (const float*)d_in[13];
  const float* W1 = (const float*)d_in[14]; const float* b1 = (const float*)d_in[15];
  const float* W2 = (const float*)d_in[16]; const float* b2 = (const float*)d_in[17];
  const float* Wf = (const float*)d_in[18]; const float* bf = (const float*)d_in[19];
  float* out = (float*)d_out;
  (void)in_sizes; (void)n_in; (void)out_size; (void)ws_size;

  char* ws = (char*)d_ws;
  size_t o = 0;
  auto alloc = [&](size_t bytes) -> size_t {
    size_t r = o; o += (bytes + 255) & ~(size_t)255; return r;
  };
  _Float16* Bfrag = (_Float16*)(ws + alloc((size_t)KPAD * CDIM * 2));
  float*    wfull = (float*)(ws + alloc((size_t)KPAD * CDIM * 4));
  float*    wq    = (float*)(ws + alloc((size_t)NREL * 2 * 4));
  float*    wk    = (float*)(ws + alloc((size_t)NREL * 2 * 4));
  float*    ew    = (float*)(ws + alloc(2 * 4));
  float*    alpha = (float*)(ws + alloc((size_t)N_EDGES * 4));
  float*    amax  = (float*)(ws + alloc((size_t)N_NODES * 4));
  float*    den   = (float*)(ws + alloc((size_t)N_NODES * 4));
  float*    sbuf  = (float*)(ws + alloc((size_t)N_NODES * KPAD * 4));
  float*    hbuf  = (float*)(ws + alloc((size_t)N_NODES * CDIM * 4));

  k_init<<<(N_NODES * KPAD + 255) / 256, 256, 0, stream>>>(amax, den, sbuf);
  k_compose_w<<<(KPAD * CDIM + 255) / 256, 256, 0, stream>>>(att, basis, wfull, Bfrag);
  k_compose_small<<<1, 128, 0, stream>>>(wfull, qv, kv, lew, ev, wq, wk, ew);
  k_edge_alpha<<<(N_EDGES + 255) / 256, 256, 0, stream>>>(x, ei, et, ea, wq, wk, ew, alpha, amax);
  k_edge_exp<<<(N_EDGES + 255) / 256, 256, 0, stream>>>(ei, alpha, amax, den);
  k_edge_scatter<<<(N_EDGES + 255) / 256, 256, 0, stream>>>(x, ei, et, alpha, den, sbuf);
  k_gemm_wmma<<<(NROWT + 7) / 8, 256, 0, stream>>>(sbuf, Bfrag, cbias, hbuf);
  k_mlp<<<(N_NODES + 63) / 64, 64, 0, stream>>>(hbuf, nty, W0, b0, W1, b1, W2, b2, Wf, bf, out);
}
